// RNNModel1_7000796692741
// MI455X (gfx1250) — matrix operand order repaired and checked by reading, not compiled, not run on hardware
//
#include <hip/hip_runtime.h>
#include <hip/hip_bf16.h>
#include <math.h>

// ---------------------------------------------------------------------------
// Problem constants
// ---------------------------------------------------------------------------
#define BB   64
#define TT   512
#define FF   784
#define HH   512
#define OUTN 10
#define G4   (4 * HH)      // 2048
#define KP   800           // F padded up to multiple of 32 (25 * 32)
#define MTOT (BB * TT)     // 32768 rows of the xz GEMM

#define STEP_BLOCKS 32     // persistent recurrence: 32 blocks x 4 waves = 128 waves
#define STEP_WAVES  4
#define WAVE_LDS_ELEMS (4 * 16 * HH)              // 32768 bf16 = 64KB per wave
#define BLOCK_LDS_BYTES (STEP_WAVES * WAVE_LDS_ELEMS * 2)  // 256KB (<320KB WGP LDS)

typedef __attribute__((ext_vector_type(16))) __bf16 v16bf;
typedef __attribute__((ext_vector_type(8)))  __bf16 v8bf;
typedef __attribute__((ext_vector_type(8)))  float  v8f;

// ---------------------------------------------------------------------------
// WMMA helpers (CDNA5: D = A(16x32 bf16) x B(32x16 bf16) + C(16x16 f32))
// ---------------------------------------------------------------------------
__device__ __forceinline__ v8f wmma_bf16(v16bf a, v16bf b, v8f c) {
    return __builtin_amdgcn_wmma_f32_16x16x32_bf16(
        /*neg_a=*/false, a, /*neg_b=*/false, b,
        /*c_mod=*/(short)0, c, /*reuse_a=*/false, /*reuse_b=*/false);
}

// A-matrix 16x32 bf16 order used here (contiguous K, same as load_tile_b):
//  lanes 0-15 : row = lane,    elems 0..15 -> K = k0+0..15
//  lanes 16-31: row = lane-16, elems 0..15 -> K = k0+16..31
__device__ __forceinline__ v16bf load_tile_a(const __bf16* base /*at (row0,0)*/,
                                             size_t row_stride, int k0, int lane) {
    const __bf16* p = base + (size_t)(lane & 15) * row_stride + (size_t)k0 + ((lane >> 4) << 4);
    v8bf lo = *(const v8bf*)p;
    v8bf hi = *(const v8bf*)(p + 8);
    v16bf r;
#pragma unroll
    for (int i = 0; i < 8; ++i) { r[i] = lo[i]; r[i + 8] = hi[i]; }
    return r;
}

// B-matrix 32x16 bf16 order used here (same K order as load_tile_a): lane holds one output column (lane&15);
// lanes 0-15 hold K=k0..k0+15, lanes 16-31 hold K=k0+16..k0+31, contiguous in K.
// bt_base points at (col0, 0) of the TRANSPOSED B (rows = output cols, ld = K dim).
__device__ __forceinline__ v16bf load_tile_b(const __bf16* bt_base,
                                             size_t row_stride, int k0, int lane) {
    const __bf16* p = bt_base + (size_t)(lane & 15) * row_stride + (size_t)k0 + ((lane >> 4) << 4);
    return *(const v16bf*)p;   // 32 bytes, 32B-aligned by construction
}

__device__ __forceinline__ float sigmoidf_(float x) { return 1.0f / (1.0f + __expf(-x)); }

// ---------------------------------------------------------------------------
// Pack / convert kernels (f32 -> bf16, transpose + zero-pad K where needed)
// ---------------------------------------------------------------------------
__global__ void pack_x_kernel(const float* __restrict__ x, __bf16* __restrict__ xb) {
    size_t idx = (size_t)blockIdx.x * blockDim.x + threadIdx.x;
    if (idx >= (size_t)MTOT * KP) return;
    int k = (int)(idx % KP);
    size_t row = idx / KP;
    float v = (k < FF) ? x[row * FF + k] : 0.0f;
    xb[idx] = (__bf16)v;
}

__global__ void pack_wt_kernel(const float* __restrict__ W, __bf16* __restrict__ wt) {
    size_t idx = (size_t)blockIdx.x * blockDim.x + threadIdx.x;
    if (idx >= (size_t)G4 * KP) return;
    int k = (int)(idx % KP);
    int n = (int)(idx / KP);
    float v = (k < FF) ? W[(size_t)k * G4 + n] : 0.0f;   // Wt[n,k] = W[k,n]
    wt[idx] = (__bf16)v;
}

__global__ void pack_ut_kernel(const float* __restrict__ U, __bf16* __restrict__ ut) {
    size_t idx = (size_t)blockIdx.x * blockDim.x + threadIdx.x;
    if (idx >= (size_t)G4 * HH) return;
    int k = (int)(idx % HH);
    int n = (int)(idx / HH);
    ut[idx] = (__bf16)U[(size_t)k * G4 + n];             // Ut[n,k] = U[k,n]
}

__global__ void init_ctr_kernel(unsigned* __restrict__ ctr) {
    if (blockIdx.x == 0 && threadIdx.x == 0) *ctr = 0u;
}

// ---------------------------------------------------------------------------
// Kernel 1: xz = x @ W + b   (M=32768, N=2048, K=800 padded), bf16 output.
// Block = 256 threads = 8 waves, computes a 128x128 output block.
// Wave (mgrp, ngrp): 2 M-tiles x 4 N-tiles = 8 wmma per K-step.
// ---------------------------------------------------------------------------
__global__ void gemm_xz_kernel(const __bf16* __restrict__ xb,
                               const __bf16* __restrict__ wt,
                               const float*  __restrict__ bias,
                               __bf16*       __restrict__ xz) {
    const int lane = threadIdx.x & 31;
    const int wave = threadIdx.x >> 5;          // 0..7
    const int mgrp = wave & 3;                  // 0..3  -> 32 rows each
    const int ngrp = wave >> 2;                 // 0..1  -> 64 cols each
    const int row0 = blockIdx.x * 128 + mgrp * 32;
    const int col0 = blockIdx.y * 128 + ngrp * 64;

    v8f acc[2][4];
#pragma unroll
    for (int nt = 0; nt < 4; ++nt) {
        float bv = bias[col0 + nt * 16 + (lane & 15)];
#pragma unroll
        for (int mt = 0; mt < 2; ++mt)
#pragma unroll
            for (int r = 0; r < 8; ++r) acc[mt][nt][r] = bv;
    }

    const __bf16* a0base = xb + (size_t)row0 * KP;
    const __bf16* a1base = xb + (size_t)(row0 + 16) * KP;

    for (int k0 = 0; k0 < KP; k0 += 32) {
        v16bf a0 = load_tile_a(a0base, KP, k0, lane);
        v16bf a1 = load_tile_a(a1base, KP, k0, lane);
#pragma unroll
        for (int nt = 0; nt < 4; ++nt) {
            v16bf bm = load_tile_b(wt + (size_t)(col0 + nt * 16) * KP, KP, k0, lane);
            acc[0][nt] = wmma_bf16(a0, bm, acc[0][nt]);
            acc[1][nt] = wmma_bf16(a1, bm, acc[1][nt]);
        }
    }

    // C/D layout: VGPR r -> row = r + 8*(lane>=16), col = lane&15
    const int rsub = (lane >> 4) << 3;
#pragma unroll
    for (int mt = 0; mt < 2; ++mt)
#pragma unroll
        for (int nt = 0; nt < 4; ++nt)
#pragma unroll
            for (int r = 0; r < 8; ++r) {
                int row = row0 + mt * 16 + r + rsub;
                int col = col0 + nt * 16 + (lane & 15);
                xz[(size_t)row * G4 + col] = (__bf16)acc[mt][nt][r];
            }
}

// ---------------------------------------------------------------------------
// Kernel 2: PERSISTENT recurrence. One launch runs all T=512 steps.
// 32 blocks x 4 waves = 128 waves; wave (mt,nt) owns the four gate tiles
// (i,f,g,o) for batch-tile mt and h-column-tile nt, so the gate nonlinearity
// is purely per-lane across the 4 accumulators.
//
// Each wave stages its step-invariant U^T slice (4 gates x 16 cols x 512 K
// = 64KB) into LDS ONCE; 4 waves x 64KB = 256KB fits CDNA5's 320KB WGP LDS.
// Steps are separated by a device-scope atomic grid barrier (monotonic
// counter, target 32*(t+1)), with fences providing cross-WGP visibility of
// h(t-1) and avoiding 512 kernel-launch overheads.
// ---------------------------------------------------------------------------
__global__ void lstm_persistent_kernel(__bf16*       __restrict__ hseq,   // [B, T, H] bf16
                                       const __bf16* __restrict__ xz,     // [B, T, 4H] bf16
                                       const __bf16* __restrict__ ut,     // [4H, H] bf16 (U^T)
                                       float*        __restrict__ cstate, // [B, H] f32
                                       unsigned*     __restrict__ ctr) {
    extern __shared__ __bf16 lds[];

    const int lane = threadIdx.x & 31;
    const int wv   = threadIdx.x >> 5;                  // 0..3
    const int gw   = blockIdx.x * STEP_WAVES + wv;      // 0..127
    const int mt   = gw & 3;        // batch tile (B=64 -> 4 tiles)
    const int nt   = gw >> 2;       // h-column tile (H=512 -> 32 tiles)
    const int row0 = mt * 16;       // batch row base
    const int col0 = nt * 16;       // h column base
    const int rsub = (lane >> 4) << 3;
    const int cl   = lane & 15;

    // ---- stage U^T slice into LDS (once) --------------------------------
    __bf16* myb = lds + wv * WAVE_LDS_ELEMS;  // [gate][col 0..15][K 0..511]
    for (int j = lane; j < WAVE_LDS_ELEMS / 8; j += 32) {   // 16B chunks
        int rowloc = j >> 6;              // 64 chunks per (gate,col) row
        int kk     = (j & 63) << 3;       // K offset
        int g      = rowloc >> 4;
        int c      = rowloc & 15;
        *(v8bf*)(myb + (size_t)j * 8) =
            *(const v8bf*)(ut + (size_t)(g * HH + col0 + c) * HH + kk);
    }
    __syncthreads();

    const size_t astr = (size_t)TT * HH;                // batch-row stride of hseq
    const size_t xzrow = (size_t)TT * G4;

    for (int t = 0; t < TT; ++t) {
        // seed accumulators with xz[b, t, gate*H + col]
        v8f acc[4];
#pragma unroll
        for (int g = 0; g < 4; ++g)
#pragma unroll
            for (int r = 0; r < 8; ++r) {
                int b   = row0 + r + rsub;
                int col = g * HH + col0 + cl;
                acc[g][r] = (float)xz[(size_t)b * xzrow + (size_t)t * G4 + col];
            }

        if (t > 0) {
            const __bf16* hprev = hseq + (size_t)(t - 1) * HH + (size_t)row0 * astr;
            for (int k0 = 0; k0 < HH; k0 += 32) {
                v16bf a = load_tile_a(hprev, astr, k0, lane);
#pragma unroll
                for (int g = 0; g < 4; ++g) {
                    // B tile from LDS: 32 contiguous bytes per lane
                    const __bf16* p = myb + ((size_t)(g * 16 + cl) * HH + k0 + ((lane >> 4) << 4));
                    v16bf bm = *(const v16bf*)p;        // ds_load_b128 x2
                    acc[g] = wmma_bf16(a, bm, acc[g]);
                }
            }
        }

        // fused gates: i=acc[0], f=acc[1], g=acc[2], o=acc[3] (Keras order)
#pragma unroll
        for (int r = 0; r < 8; ++r) {
            int b   = row0 + r + rsub;
            int col = col0 + cl;
            float gi = sigmoidf_(acc[0][r]);
            float gf = sigmoidf_(acc[1][r]);
            float gg = tanhf(acc[2][r]);
            float go = sigmoidf_(acc[3][r]);
            size_t cidx = (size_t)b * HH + col;
            float c = (t == 0) ? 0.0f : cstate[cidx];
            c = gf * c + gi * gg;
            cstate[cidx] = c;
            float h = go * tanhf(c);
            hseq[(size_t)b * astr + (size_t)t * HH + col] = (__bf16)h;
        }

        // ---- grid-wide barrier between steps ----------------------------
        __threadfence();                        // release h(t), c
        __syncthreads();
        if (threadIdx.x == 0) {
            atomicAdd(ctr, 1u);
            unsigned target = (unsigned)(STEP_BLOCKS * (t + 1));
            while (__hip_atomic_load(ctr, __ATOMIC_RELAXED, __HIP_MEMORY_SCOPE_AGENT) < target) {
                __builtin_amdgcn_s_sleep(2);
            }
        }
        __syncthreads();
        __threadfence();                        // acquire other WGPs' h(t)
    }
}

// ---------------------------------------------------------------------------
// Kernel 3: out = hseq @ Wd + bd   (K=512, N=10) — memory-bound, plain VALU.
// One thread per (b,t) row, vectorized bf16 loads of h.
// ---------------------------------------------------------------------------
__global__ void proj_kernel(const __bf16* __restrict__ hseq,
                            const float*  __restrict__ wd,
                            const float*  __restrict__ bd,
                            float*        __restrict__ out) {
    int row = blockIdx.x * blockDim.x + threadIdx.x;
    if (row >= MTOT) return;
    const __bf16* hp = hseq + (size_t)row * HH;   // row = b*T + t matches [B,T,H]
    float acc[OUTN];
#pragma unroll
    for (int o = 0; o < OUTN; ++o) acc[o] = bd[o];
    for (int k = 0; k < HH; k += 8) {
        v8bf hv = *(const v8bf*)(hp + k);
#pragma unroll
        for (int j = 0; j < 8; ++j) {
            float h = (float)hv[j];
            const float* wrow = wd + (size_t)(k + j) * OUTN;
#pragma unroll
            for (int o = 0; o < OUTN; ++o) acc[o] = fmaf(h, wrow[o], acc[o]);
        }
    }
    float* op = out + (size_t)row * OUTN;
#pragma unroll
    for (int o = 0; o < OUTN; ++o) op[o] = acc[o];
}

// ---------------------------------------------------------------------------
// Launch
// ---------------------------------------------------------------------------
extern "C" void kernel_launch(void* const* d_in, const int* in_sizes, int n_in,
                              void* d_out, int out_size, void* d_ws, size_t ws_size,
                              hipStream_t stream) {
    const float* x  = (const float*)d_in[0];  // [B,T,F]
    const float* W  = (const float*)d_in[1];  // [F,4H]
    const float* U  = (const float*)d_in[2];  // [H,4H]
    const float* b  = (const float*)d_in[3];  // [4H]
    const float* Wd = (const float*)d_in[4];  // [H,OUT]
    const float* bd = (const float*)d_in[5];  // [OUT]
    float* out = (float*)d_out;

    // workspace layout (all offsets 256B aligned)
    char* ws = (char*)d_ws;
    size_t off = 0;
    __bf16* xb = (__bf16*)(ws + off); off += (size_t)MTOT * KP * 2;        // 52.4 MB
    __bf16* wt = (__bf16*)(ws + off); off += (size_t)G4 * KP * 2;          //  3.3 MB
    __bf16* ut = (__bf16*)(ws + off); off += (size_t)G4 * HH * 2;          //  2.1 MB
    __bf16* xz = (__bf16*)(ws + off); off += (size_t)MTOT * G4 * 2;        // 134 MB
    __bf16* hs = (__bf16*)(ws + off); off += (size_t)BB * TT * HH * 2;     // 33.6 MB
    float*  cs = (float*)(ws + off);  off += (size_t)BB * HH * 4;          // 0.13 MB
    unsigned* ctr = (unsigned*)(ws + off); off += 256;

    // pack / convert + barrier-counter init
    {
        size_t nx = (size_t)MTOT * KP;
        pack_x_kernel<<<(unsigned)((nx + 255) / 256), 256, 0, stream>>>(x, xb);
        size_t nw = (size_t)G4 * KP;
        pack_wt_kernel<<<(unsigned)((nw + 255) / 256), 256, 0, stream>>>(W, wt);
        size_t nu = (size_t)G4 * HH;
        pack_ut_kernel<<<(unsigned)((nu + 255) / 256), 256, 0, stream>>>(U, ut);
        init_ctr_kernel<<<1, 32, 0, stream>>>(ctr);
    }

    // xz = x @ W + b : grid (32768/128, 2048/128)
    gemm_xz_kernel<<<dim3(MTOT / 128, G4 / 128), 256, 0, stream>>>(xb, wt, b, xz);

    // persistent recurrence: all 512 steps in one launch
    lstm_persistent_kernel<<<STEP_BLOCKS, STEP_WAVES * 32, BLOCK_LDS_BYTES, stream>>>(
        hs, xz, ut, cs, ctr);

    // output projection
    proj_kernel<<<(MTOT + 255) / 256, 256, 0, stream>>>(hs, Wd, bd, out);
}
